// BifurcationAttentionModule_85856396247184
// MI455X (gfx1250) — matrix operand order repaired and checked by reading, not compiled, not run on hardware
//
#include <hip/hip_runtime.h>

// ---------------------------------------------------------------------------
// CDNA5 (gfx1250) fused GNN bifurcation kernel.
//   - matrix math: v_wmma_f32_16x16x32_bf16 (fp32 accumulate)
//   - ctx gather: global_load_async_to_lds_b128 (ASYNCcnt) into row-major LDS
// ---------------------------------------------------------------------------

typedef __attribute__((ext_vector_type(16))) __bf16 v16bf;
typedef __attribute__((ext_vector_type(8)))  __bf16 v8bf;
typedef __attribute__((ext_vector_type(8)))  float  v8f;

#define WG_ROWS 32      // nodes per workgroup (2 row tiles of 16)
#define FDIM    128
#define K1      384     // 3F
#define N1      256     // 2F
#define NKT1    12      // K1/32
#define NNT1    16      // N1/16
#define NKT2    8       // 256/32
#define NNT2    8       // 128/16
#define NKT3    4       // 128/32
#define NNT3    4       // 64/16

// padded row strides (elements); all keep 16-byte alignment per row
#define CTX_STRIDE 392  // 384 + 8  (784 B/row)
#define H_STRIDE   264  // 256 + 8  (528 B/row)
#define P_STRIDE   132  // fp32, 128 + 4 (528 B/row)
#define A3_STRIDE  136  // 128 + 8  (272 B/row)

// Load a 16x32 bf16 A-fragment from a row-major LDS matrix.
// ISA layout: lane L -> M = L%16, g = L/16; VGPR0..3 hold K = g*8 + 0..7,
// VGPR4..7 hold K = 16 + g*8 + 0..7  => two contiguous 16-byte chunks.
__device__ __forceinline__ v16bf load_a_frag(const __bf16* buf, int row,
                                             int stride, int kBase, int lane) {
    const __bf16* p = buf + row * stride + kBase + ((lane >> 4) << 3);
    v8bf lo = *(const v8bf*)p;          // K = kBase + g*8 + 0..7
    v8bf hi = *(const v8bf*)(p + 16);   // K = kBase + 16 + g*8 + 0..7
    return __builtin_shufflevector(lo, hi, 0, 1, 2, 3, 4, 5, 6, 7,
                                            8, 9, 10, 11, 12, 13, 14, 15);
}

// ---------------------------------------------------------------------------
// Graph prep: two smallest distinct neighbors via atomicMin
// ---------------------------------------------------------------------------
__global__ void init_min_kernel(int* __restrict__ min1, int* __restrict__ min2,
                                int n, int sentinel) {
    int i = blockIdx.x * blockDim.x + threadIdx.x;
    if (i < n) { min1[i] = sentinel; min2[i] = sentinel; }
}

__global__ void edge_min1_kernel(const int* __restrict__ ei, int e,
                                 int* __restrict__ min1) {
    int i = blockIdx.x * blockDim.x + threadIdx.x;
    if (i >= e) return;
    int u = ei[i], v = ei[e + i];
    if (u != v) {
        atomicMin(&min1[u], v);
        atomicMin(&min1[v], u);
    }
}

__global__ void edge_min2_kernel(const int* __restrict__ ei, int e,
                                 const int* __restrict__ min1,
                                 int* __restrict__ min2) {
    int i = blockIdx.x * blockDim.x + threadIdx.x;
    if (i >= e) return;
    int u = ei[i], v = ei[e + i];
    if (u != v) {
        if (v != min1[u]) atomicMin(&min2[u], v);
        if (u != min1[v]) atomicMin(&min2[v], u);
    }
}

// ---------------------------------------------------------------------------
// node_features fp32 -> bf16 staging table (async-load friendly; fits in L2)
// ---------------------------------------------------------------------------
__global__ void conv_bf16_kernel(const float* __restrict__ src,
                                 __bf16* __restrict__ dst, int total) {
    int i = blockIdx.x * blockDim.x + threadIdx.x;
    if (i < total) dst[i] = (__bf16)src[i];
}

// ---------------------------------------------------------------------------
// Pack fp32 weight [K x Nout] into bf16 WMMA B-fragment order:
// idx = ((kTile*nNT + nTile)*32 + lane)*16 + e
// value = w[kTile*32 + (lane/16)*8 + (e%8) + (e/8)*16][nTile*16 + lane%16]
// ---------------------------------------------------------------------------
__global__ void pack_b_kernel(const float* __restrict__ src,
                              __bf16* __restrict__ dst, int K, int Nout) {
    int idx = blockIdx.x * blockDim.x + threadIdx.x;
    int total = K * Nout;
    if (idx >= total) return;
    int e     = idx & 15;
    int lane  = (idx >> 4) & 31;
    int tile  = idx >> 9;
    int nNT   = Nout >> 4;
    int kTile = tile / nNT;
    int nTile = tile - kTile * nNT;
    int k = kTile * 32 + ((lane >> 4) << 3) + (e & 7) + ((e >> 3) << 4);
    int c = nTile * 16 + (lane & 15);
    dst[idx] = (__bf16)src[k * Nout + c];
}

// ---------------------------------------------------------------------------
// Fused: async ctx gather -> L1 GEMM+relu -> L2 GEMM -> LN -> L3 GEMM+relu ->
//        murray dot+sigmoid -> masked output
// ---------------------------------------------------------------------------
__global__ void __launch_bounds__(256)
fused_gnn_kernel(const float* __restrict__ nf, const __bf16* __restrict__ nf16,
                 const int* __restrict__ min1, const int* __restrict__ min2,
                 const int* __restrict__ node_types,
                 const __bf16* __restrict__ w1p, const float* __restrict__ b1,
                 const __bf16* __restrict__ w2p, const float* __restrict__ b2,
                 const float* __restrict__ ln_g, const float* __restrict__ ln_b,
                 const __bf16* __restrict__ mw1p, const float* __restrict__ mb1,
                 const float* __restrict__ mw2, const float* __restrict__ mb2,
                 float* __restrict__ out_updated, float* __restrict__ out_murray,
                 int n) {
    // Region A (25.6 KB): ctx row-major (layer1), later proc fp32 + A3 bf16
    __shared__ __align__(32) unsigned char smemA[WG_ROWS * P_STRIDE * 4 +
                                                 WG_ROWS * A3_STRIDE * 2]; // 25600
    // Region B (16.9 KB): h row-major (layer2 input)
    __shared__ __align__(32) unsigned char smemB[WG_ROWS * H_STRIDE * 2];
    __shared__ int   sN0[WG_ROWS], sN1[WG_ROWS], sUpd[WG_ROWS];
    __shared__ int   sIsBif[WG_ROWS], sHasTwo[WG_ROWS];
    __shared__ float sMu[WG_ROWS], sRstd[WG_ROWS], sMacc[WG_ROWS];

    __bf16* Ctx   = (__bf16*)smemA;                              // [32][392]
    float*  procB = (float*)smemA;                               // [32][132]
    __bf16* A3    = (__bf16*)(smemA + WG_ROWS * P_STRIDE * 4);   // [32][136]
    __bf16* Hbuf  = (__bf16*)smemB;                              // [32][264]

    const int tid     = threadIdx.x;
    const int lane    = tid & 31;
    const int w       = tid >> 5;
    const int base    = blockIdx.x * WG_ROWS;
    const int rowTile = w & 1;
    const int nGroup  = w >> 1;

    if (tid == 0) {
        __builtin_prefetch((const void*)w1p, 0, 1);   // global_prefetch_b8
        __builtin_prefetch((const void*)w2p, 0, 1);
        __builtin_prefetch((const void*)mw1p, 0, 1);
    }

    // ---- per-row metadata -------------------------------------------------
    if (tid < WG_ROWS) {
        int node = base + tid;
        int n0 = 0, n1 = 0, hb = 0, ht = 0;
        if (node < n) {
            int m1 = min1[node], m2 = min2[node];
            ht = (m1 < n) && (m2 < n);
            n0 = (m1 < n) ? m1 : n - 1;
            n1 = (m2 < n) ? m2 : n - 1;
            hb = (node_types[node] == 1);
        }
        sN0[tid] = n0; sN1[tid] = n1;
        sIsBif[tid] = hb; sHasTwo[tid] = ht;
        sUpd[tid] = hb && ht;
        sMacc[tid] = 0.0f;
    }
    __syncthreads();

    // ---- async gather ctx = [self | nbr0 | nbr1] (48 x 16B chunks / row) --
    // 32 rows * 48 chunks = 1536 = 256 threads * 6 -> divergence-free loop.
    for (int c = tid; c < WG_ROWS * 48; c += 256) {
        int r  = c / 48;
        int q  = c - r * 48;
        int srcSel = q >> 4;          // 0: self, 1: nbr0, 2: nbr1
        int fc     = q & 15;          // 8-feature chunk within the source row
        int node   = base + r;
        int self   = (node < n) ? node : n - 1;
        int src    = (srcSel == 0) ? self : ((srcSel == 1) ? sN0[r] : sN1[r]);
        unsigned long long gaddr =
            (unsigned long long)(const void*)(nf16 + (size_t)src * FDIM + fc * 8);
        unsigned ldsAddr =
            (unsigned)(unsigned long long)(void*)&Ctx[r * CTX_STRIDE + q * 8];
        asm volatile("global_load_async_to_lds_b128 %0, %1, off"
                     :: "v"(ldsAddr), "v"(gaddr)
                     : "memory");
    }
    asm volatile("s_wait_asynccnt 0x0" ::: "memory");
    __syncthreads();

    // ---- layer 1: [32x384] @ [384x256] ------------------------------------
    const int aRow = rowTile * 16 + (lane & 15);
    v8f acc1[4];
#pragma unroll
    for (int t = 0; t < 4; ++t) acc1[t] = (v8f){0, 0, 0, 0, 0, 0, 0, 0};
    for (int kt = 0; kt < NKT1; ++kt) {
        v16bf a = load_a_frag(Ctx, aRow, CTX_STRIDE, kt * 32, lane);
#pragma unroll
        for (int t = 0; t < 4; ++t) {
            int nT = nGroup * 4 + t;
            v16bf b = *(const v16bf*)&w1p[((kt * NNT1 + nT) * 32 + lane) * 16];
            acc1[t] = __builtin_amdgcn_wmma_f32_16x16x32_bf16(
                false, a, false, b, (short)0, acc1[t], false, false);
        }
    }
    // bias + relu -> h (row-major bf16)
#pragma unroll
    for (int t = 0; t < 4; ++t) {
        int col = (nGroup * 4 + t) * 16 + (lane & 15);
        float bv = b1[col];
#pragma unroll
        for (int i = 0; i < 8; ++i) {
            float v = acc1[t][i] + bv;
            v = v > 0.0f ? v : 0.0f;
            int r = rowTile * 16 + i + ((lane >> 4) << 3);
            Hbuf[r * H_STRIDE + col] = (__bf16)v;
        }
    }
    __syncthreads();

    // ---- layer 2: [32x256] @ [256x128] ------------------------------------
    v8f acc2[2];
    acc2[0] = (v8f){0, 0, 0, 0, 0, 0, 0, 0};
    acc2[1] = (v8f){0, 0, 0, 0, 0, 0, 0, 0};
    for (int kt = 0; kt < NKT2; ++kt) {
        v16bf a = load_a_frag(Hbuf, aRow, H_STRIDE, kt * 32, lane);
#pragma unroll
        for (int t = 0; t < 2; ++t) {
            int nT = nGroup * 2 + t;
            v16bf b = *(const v16bf*)&w2p[((kt * NNT2 + nT) * 32 + lane) * 16];
            acc2[t] = __builtin_amdgcn_wmma_f32_16x16x32_bf16(
                false, a, false, b, (short)0, acc2[t], false, false);
        }
    }
#pragma unroll
    for (int t = 0; t < 2; ++t) {
        int col = (nGroup * 2 + t) * 16 + (lane & 15);
        float bv = b2[col];
#pragma unroll
        for (int i = 0; i < 8; ++i) {
            int r = rowTile * 16 + i + ((lane >> 4) << 3);
            procB[r * P_STRIDE + col] = acc2[t][i] + bv;  // region A reuse (post-barrier)
        }
    }
    __syncthreads();

    // ---- layernorm --------------------------------------------------------
    if (tid < WG_ROWS) {
        const float* row = procB + tid * P_STRIDE;
        float s = 0.0f;
        for (int j = 0; j < FDIM; ++j) s += row[j];
        float mu = s * (1.0f / FDIM);
        float v2 = 0.0f;
        for (int j = 0; j < FDIM; ++j) {
            float d = row[j] - mu;
            v2 += d * d;
        }
        sMu[tid]   = mu;
        sRstd[tid] = rsqrtf(v2 * (1.0f / FDIM) + 1e-5f);
    }
    __syncthreads();
    for (int idx = tid; idx < WG_ROWS * FDIM; idx += 256) {
        int r = idx >> 7, f = idx & 127;
        float xv = (procB[r * P_STRIDE + f] - sMu[r]) * sRstd[r] * ln_g[f] + ln_b[f];
        procB[r * P_STRIDE + f] = xv;                 // keep fp32 for output
        A3[r * A3_STRIDE + f]   = (__bf16)xv;         // bf16 copy for layer 3
    }
    __syncthreads();

    // ---- layer 3: [32x128] @ [128x64], relu, dot mw2 ----------------------
    v8f acc3 = (v8f){0, 0, 0, 0, 0, 0, 0, 0};
    for (int kt = 0; kt < NKT3; ++kt) {
        v16bf a = load_a_frag(A3, aRow, A3_STRIDE, kt * 32, lane);
        v16bf b = *(const v16bf*)&mw1p[((kt * NNT3 + nGroup) * 32 + lane) * 16];
        acc3 = __builtin_amdgcn_wmma_f32_16x16x32_bf16(
            false, a, false, b, (short)0, acc3, false, false);
    }
    {
        int col = nGroup * 16 + (lane & 15);
        float bv = mb1[col];
        float wv = mw2[col];
#pragma unroll
        for (int i = 0; i < 8; ++i) {
            float z = acc3[i] + bv;
            z = z > 0.0f ? z : 0.0f;
            int r = rowTile * 16 + i + ((lane >> 4) << 3);
            atomicAdd(&sMacc[r], z * wv);             // ds_add_f32
        }
    }
    __syncthreads();

    // ---- murray score + masked output -------------------------------------
    if (tid < WG_ROWS) {
        int node = base + tid;
        if (node < n) {
            float mv = 0.0f;
            if (sIsBif[tid]) {
                if (sHasTwo[tid]) {
                    float x = sMacc[tid] + mb2[0];
                    mv = 1.0f / (1.0f + expf(-x));
                } else {
                    mv = 0.5f;
                }
            }
            out_murray[node] = mv;
        }
    }
    for (int idx = tid; idx < WG_ROWS * FDIM; idx += 256) {
        int r = idx >> 7, f = idx & 127;
        int node = base + r;
        if (node < n) {
            float vout = sUpd[r] ? procB[r * P_STRIDE + f] : nf[node * FDIM + f];
            out_updated[node * FDIM + f] = vout;
        }
    }
}

// ---------------------------------------------------------------------------
// Launch
// ---------------------------------------------------------------------------
extern "C" void kernel_launch(void* const* d_in, const int* in_sizes, int n_in,
                              void* d_out, int out_size, void* d_ws, size_t ws_size,
                              hipStream_t stream) {
    const float* nf   = (const float*)d_in[0];
    const int*   ei   = (const int*)  d_in[1];
    const int*   nty  = (const int*)  d_in[2];
    const float* w1   = (const float*)d_in[3];
    const float* b1   = (const float*)d_in[4];
    const float* w2   = (const float*)d_in[5];
    const float* b2   = (const float*)d_in[6];
    const float* ln_g = (const float*)d_in[7];
    const float* ln_b = (const float*)d_in[8];
    const float* mw1  = (const float*)d_in[9];
    const float* mb1  = (const float*)d_in[10];
    const float* mw2  = (const float*)d_in[11];
    const float* mb2  = (const float*)d_in[12];

    const int N = in_sizes[0] / FDIM;
    const int E = in_sizes[1] / 2;
    const int NFTOT = N * FDIM;

    // workspace carve (keeps every bf16 block 32B-aligned)
    char* ws = (char*)d_ws;
    __bf16* nf16 = (__bf16*)ws;      ws += (size_t)NFTOT * sizeof(__bf16);
    int* min1 = (int*)ws;            ws += (size_t)N * sizeof(int);
    int* min2 = (int*)ws;            ws += (size_t)N * sizeof(int);
    __bf16* w1p  = (__bf16*)ws;      ws += (size_t)K1 * N1 * sizeof(__bf16);
    __bf16* w2p  = (__bf16*)ws;      ws += (size_t)256 * 128 * sizeof(__bf16);
    __bf16* mw1p = (__bf16*)ws;      ws += (size_t)128 * 64 * sizeof(__bf16);

    float* out_updated = (float*)d_out;
    float* out_murray  = (float*)d_out + (size_t)N * FDIM;

    init_min_kernel<<<(N + 255) / 256, 256, 0, stream>>>(min1, min2, N, N);
    edge_min1_kernel<<<(E + 255) / 256, 256, 0, stream>>>(ei, E, min1);
    edge_min2_kernel<<<(E + 255) / 256, 256, 0, stream>>>(ei, E, min1, min2);

    conv_bf16_kernel<<<(NFTOT + 255) / 256, 256, 0, stream>>>(nf, nf16, NFTOT);

    pack_b_kernel<<<(K1 * N1 + 255) / 256, 256, 0, stream>>>(w1, w1p, K1, N1);
    pack_b_kernel<<<(256 * 128 + 255) / 256, 256, 0, stream>>>(w2, w2p, 256, 128);
    pack_b_kernel<<<(128 * 64 + 255) / 256, 256, 0, stream>>>(mw1, mw1p, 128, 64);

    fused_gnn_kernel<<<(N + WG_ROWS - 1) / WG_ROWS, 256, 0, stream>>>(
        nf, nf16, min1, min2, nty,
        w1p, b1, w2p, b2, ln_g, ln_b, mw1p, mb1, mw2, mb2,
        out_updated, out_murray, N);
}
